// VEM_26242250179294
// MI455X (gfx1250) — compile-verified, matrix operand order checked by reading, not run on hardware
//
#include <hip/hip_runtime.h>
#include <hip/hip_bf16.h>
#include <math.h>

#define TEMP_INV 20.0f   // 1/0.05
#define ETA 10
#define MDIM 8192
#define NDIM 2048
#define HDIM 1024

typedef __attribute__((ext_vector_type(16))) __bf16 v16bf;
typedef __attribute__((ext_vector_type(8)))  __bf16 v8bf;
typedef __attribute__((ext_vector_type(8)))  float  v8f;

// ---------------------------------------------------------------------------
// Tiled bf16 WMMA GEMM: C[M,N] = scale * opA(A) * opB(B), fp32 in/out.
//  AT=false: A is [M,K] row-major (element (m,k) = A[m*lda+k])
//  AT=true : A stored [K,M] row-major (element (m,k) = A[k*lda+m])
//  BT=false: B is [K,N] row-major   BT=true: B stored [N,K] row-major
// Block tile 128x128, BK=32, 256 threads (8 waves, 4(M) x 2(N)),
// wave tile 32x64 = 2x4 v_wmma_f32_16x16x32_bf16.
// Double-buffered LDS (one barrier per K-step); next tile's global loads are
// issued before the current step's WMMAs so vmem latency hides under matrix
// core work.
// ---------------------------------------------------------------------------
template<bool AT, bool BT>
__global__ __launch_bounds__(256)
void gemm_bf16_wmma(const float* __restrict__ A, const float* __restrict__ B,
                    float* __restrict__ C, int K, int lda, int ldb, int ldc,
                    float scale)
{
    __shared__ __align__(16) __bf16 At[2][128][32];   // [buf][m][k]
    __shared__ __align__(16) __bf16 Bt[2][128][32];   // [buf][n][k] (transposed)

    const int t      = threadIdx.x;
    const int blockM = blockIdx.y * 128;
    const int blockN = blockIdx.x * 128;

    const int lane = t & 31;
    const int wid  = t >> 5;
    const int wm   = (wid & 3) * 32;   // wave M offset within block tile
    const int wn   = (wid >> 2) * 64;  // wave N offset within block tile
    const int half = lane >> 4;        // K-half select per ISA layout
    const int r    = lane & 15;        // row/col within fragment

    v8f acc[2][4];
#pragma unroll
    for (int mi = 0; mi < 2; ++mi)
#pragma unroll
        for (int ni = 0; ni < 4; ++ni)
#pragma unroll
            for (int e = 0; e < 8; ++e) acc[mi][ni][e] = 0.0f;

    const int sr = t >> 1;           // staging row 0..127
    const int sk = (t & 1) * 16;     // staging k offset 0/16
    const int gm = blockM + sr;
    const int gn = blockN + sr;

    float aReg[16], bReg[16];

    // prefetch of tile kt into registers
    auto load_global = [&](int kt) {
        const int kbase = kt * 32 + sk;
#pragma unroll
        for (int j = 0; j < 16; ++j) {
            const int gk = kbase + j;
            aReg[j] = AT ? A[(size_t)gk * lda + gm] : A[(size_t)gm * lda + gk];
        }
#pragma unroll
        for (int j = 0; j < 16; ++j) {
            const int gk = kbase + j;
            bReg[j] = BT ? B[(size_t)gn * ldb + gk] : B[(size_t)gk * ldb + gn];
        }
    };
    // convert f32 -> bf16 and store registers to LDS buffer `buf`
    auto store_lds = [&](int buf) {
#pragma unroll
        for (int j = 0; j < 16; ++j) At[buf][sr][sk + j] = (__bf16)aReg[j];
#pragma unroll
        for (int j = 0; j < 16; ++j) Bt[buf][sr][sk + j] = (__bf16)bReg[j];
    };

    const int nK = K >> 5;
    load_global(0);
    store_lds(0);
    __syncthreads();

    for (int kt = 0; kt < nK; ++kt) {
        const int cur = kt & 1;

        // issue next tile's global loads first (overlap with WMMA below)
        if (kt + 1 < nK) load_global(kt + 1);

        // ---- load fragments per 16-bit A-matrix VGPR layout ----
        v16bf af[2], bfrag[4];
#pragma unroll
        for (int mi = 0; mi < 2; ++mi) {
            const __bf16* ap = &At[cur][wm + mi * 16 + r][0];
            const v8bf lo = *(const v8bf*)(ap + half * 8);       // K = half*8 + 0..7
            const v8bf hi = *(const v8bf*)(ap + 16 + half * 8);  // K = 16 + half*8 + 0..7
#pragma unroll
            for (int i = 0; i < 8; ++i) { af[mi][i] = lo[i]; af[mi][i + 8] = hi[i]; }
        }
#pragma unroll
        for (int ni = 0; ni < 4; ++ni) {
            const __bf16* bp = &Bt[cur][wn + ni * 16 + r][0];
            const v8bf lo = *(const v8bf*)(bp + half * 8);
            const v8bf hi = *(const v8bf*)(bp + 16 + half * 8);
#pragma unroll
            for (int i = 0; i < 8; ++i) { bfrag[ni][i] = lo[i]; bfrag[ni][i + 8] = hi[i]; }
        }

#pragma unroll
        for (int mi = 0; mi < 2; ++mi)
#pragma unroll
            for (int ni = 0; ni < 4; ++ni)
                acc[mi][ni] = __builtin_amdgcn_wmma_f32_16x16x32_bf16(
                    false, af[mi], false, bfrag[ni], (short)0, acc[mi][ni],
                    false, false);

        // fill the other buffer; single barrier per K-step
        if (kt + 1 < nK) {
            store_lds(cur ^ 1);
            __syncthreads();
        }
    }

    // ---- epilogue: C layout VGPR rr -> M = half*8 + rr, lane&15 -> N ----
#pragma unroll
    for (int mi = 0; mi < 2; ++mi) {
        const int row0 = blockM + wm + mi * 16 + half * 8;
#pragma unroll
        for (int ni = 0; ni < 4; ++ni) {
            const int col = blockN + wn + ni * 16 + r;
#pragma unroll
            for (int rr = 0; rr < 8; ++rr)
                C[(size_t)(row0 + rr) * ldc + col] = scale * acc[mi][ni][rr];
        }
    }
}

// ---------------------------------------------------------------------------
// Block reductions (blockDim.x == 256)
// ---------------------------------------------------------------------------
__device__ __forceinline__ float breduce_sum(float v, float* s) {
    const int t = threadIdx.x;
    s[t] = v; __syncthreads();
#pragma unroll
    for (int off = 128; off > 0; off >>= 1) {
        if (t < off) s[t] += s[t + off];
        __syncthreads();
    }
    const float r = s[0]; __syncthreads();
    return r;
}
__device__ __forceinline__ float breduce_max(float v, float* s) {
    const int t = threadIdx.x;
    s[t] = v; __syncthreads();
#pragma unroll
    for (int off = 128; off > 0; off >>= 1) {
        if (t < off) s[t] = fmaxf(s[t], s[t + off]);
        __syncthreads();
    }
    const float r = s[0]; __syncthreads();
    return r;
}

// ---------------------------------------------------------------------------
// Normalize q rows -> qn, and write cos_sim column into d_out col 0.
// One block (256 thr) per row, H=1024 -> 4 elems/thread.
// ---------------------------------------------------------------------------
__global__ __launch_bounds__(256)
void norm_qk_kernel(const float* __restrict__ q, const float* __restrict__ k,
                    float* __restrict__ qn, float* __restrict__ out, int ldo)
{
    __shared__ float sred[256];
    const int t = threadIdx.x;
    const size_t base = (size_t)blockIdx.x * HDIM;

    float qv[4], kv[4];
    float qq = 0.f, kk = 0.f, qk = 0.f;
#pragma unroll
    for (int j = 0; j < 4; ++j) {
        const int idx = t + j * 256;
        qv[j] = q[base + idx];
        kv[j] = k[base + idx];
        qq += qv[j] * qv[j];
        kk += kv[j] * kv[j];
        qk += qv[j] * kv[j];
    }
    qq = breduce_sum(qq, sred);
    kk = breduce_sum(kk, sred);
    qk = breduce_sum(qk, sred);

    const float invq = 1.0f / fmaxf(sqrtf(qq), 1e-12f);
    const float invk = 1.0f / fmaxf(sqrtf(kk), 1e-12f);
#pragma unroll
    for (int j = 0; j < 4; ++j)
        qn[base + t + j * 256] = qv[j] * invq;
    if (t == 0)
        out[(size_t)blockIdx.x * ldo] = qk * invq * invk * TEMP_INV;
}

__global__ __launch_bounds__(256)
void copy_kernel(const float* __restrict__ src, float* __restrict__ dst)
{
    const size_t i = (size_t)blockIdx.x * 256 + threadIdx.x;
    dst[i] = src[i];
}

// ---------------------------------------------------------------------------
// Row softmax of L/TEMP (in place, fp32) + meanv[m] = mean(L_norm * L).
// One block per row of M=8192; N=2048 -> 8 elems/thread.
// ---------------------------------------------------------------------------
__global__ __launch_bounds__(256)
void softmax_kernel(float* __restrict__ L, float* __restrict__ meanv)
{
    __shared__ float sred[256];
    const int t = threadIdx.x;
    const size_t base = (size_t)blockIdx.x * NDIM;

    float x[8], p[8];
    float mx = -3.402823466e+38f;
#pragma unroll
    for (int j = 0; j < 8; ++j) {
        x[j] = L[base + t + j * 256];
        mx = fmaxf(mx, x[j] * TEMP_INV);
    }
    mx = breduce_max(mx, sred);

    float se = 0.f, sd = 0.f;
#pragma unroll
    for (int j = 0; j < 8; ++j) {
        p[j] = __expf(x[j] * TEMP_INV - mx);
        se += p[j];
        sd += p[j] * x[j];
    }
    se = breduce_sum(se, sred);
    sd = breduce_sum(sd, sred);

    const float inv = 1.0f / se;
#pragma unroll
    for (int j = 0; j < 8; ++j)
        L[base + t + j * 256] = p[j] * inv;
    if (t == 0)
        meanv[blockIdx.x] = sd * inv * (1.0f / (float)NDIM);
}

// ---------------------------------------------------------------------------
// dB1 = dB1/N - meanv[m] * B   (elementwise over M*H)
// ---------------------------------------------------------------------------
__global__ __launch_bounds__(256)
void db1_fix_kernel(float* __restrict__ dB1, const float* __restrict__ Bf,
                    const float* __restrict__ meanv, float invN)
{
    const size_t i = (size_t)blockIdx.x * 256 + threadIdx.x;
    const int m = (int)(i / HDIM);
    dB1[i] = dB1[i] * invN - meanv[m] * Bf[i];
}

// ---------------------------------------------------------------------------
// Langevin update + row l2norm:
//  v = B + c*(P/M + B) + c*dB1 + s*Q ;  B = v / max(||v||, eps)
// One block per row of M; H=1024 -> 4 elems/thread.
// ---------------------------------------------------------------------------
__global__ __launch_bounds__(256)
void update_kernel(float* __restrict__ Bf, const float* __restrict__ dB1,
                   const float* __restrict__ P, const float* __restrict__ Q,
                   float c, float s, float invM)
{
    __shared__ float sred[256];
    const int t = threadIdx.x;
    const size_t base = (size_t)blockIdx.x * HDIM;

    float v[4];
    float ss = 0.f;
#pragma unroll
    for (int j = 0; j < 4; ++j) {
        const int idx = t + j * 256;
        const float b  = Bf[base + idx];
        const float d1 = dB1[base + idx];
        const float db = P[base + idx] * invM + b;   // delta_B
        v[j] = b + c * db + c * d1 + s * Q[base + idx];
        ss += v[j] * v[j];
    }
    ss = breduce_sum(ss, sred);
    const float inv = 1.0f / fmaxf(sqrtf(ss), 1e-12f);
#pragma unroll
    for (int j = 0; j < 4; ++j)
        Bf[base + t + j * 256] = v[j] * inv;
}

// ---------------------------------------------------------------------------
extern "C" void kernel_launch(void* const* d_in, const int* in_sizes, int n_in,
                              void* d_out, int out_size, void* d_ws, size_t ws_size,
                              hipStream_t stream)
{
    const float* q     = (const float*)d_in[0];   // [N,H]
    const float* k     = (const float*)d_in[1];   // [N,H]
    const float* Binit = (const float*)d_in[2];   // [M,H]
    const float* noise = (const float*)d_in[3];   // [ETA,M,H]
    float* out = (float*)d_out;                   // [N, 1+M]

    // workspace layout (fp32): everything lives in the 192MB L2 working set
    float* Bf    = (float*)d_ws;                        // M*H
    float* qn    = Bf   + (size_t)MDIM * HDIM;          // N*H
    float* Lb    = qn   + (size_t)NDIM * HDIM;          // M*N (also hosts P)
    float* dB1   = Lb   + (size_t)MDIM * NDIM;          // M*H
    float* G     = dB1  + (size_t)MDIM * HDIM;          // H*H
    float* meanv = G    + (size_t)HDIM * HDIM;          // M
    float* P     = Lb;  // L is dead after dB1 GEMM; reuse its space for P

    // q/k normalization + cos_sim column (d_out col 0)
    norm_qk_kernel<<<NDIM, 256, 0, stream>>>(q, k, qn, out, MDIM + 1);
    // B <- B_init
    copy_kernel<<<(MDIM * HDIM) / 256, 256, 0, stream>>>(Binit, Bf);

    for (int i = 0; i < ETA; ++i) {
        const float tt = (float)(i + 1);
        const float c  = 0.5f / tt;            // ALPHA * 0.5 / t
        const float sn = sqrtf(2.0f / tt);     // sqrt(2*ALPHA/t)

        // L[M,N] = B[M,H] @ qn^T  (NT)
        gemm_bf16_wmma<false, true><<<dim3(NDIM / 128, MDIM / 128), 256, 0, stream>>>(
            Bf, qn, Lb, HDIM, HDIM, HDIM, NDIM, 1.0f);

        // softmax rows in place + meanv = mean(L_norm*L)
        softmax_kernel<<<MDIM, 256, 0, stream>>>(Lb, meanv);

        // dB1raw[M,H] = L_norm[M,N] @ qn[N,H]  (NN)
        gemm_bf16_wmma<false, false><<<dim3(HDIM / 128, MDIM / 128), 256, 0, stream>>>(
            Lb, qn, dB1, NDIM, NDIM, HDIM, HDIM, 1.0f);

        // dB1 = dB1raw/N - meanv*B
        db1_fix_kernel<<<(MDIM * HDIM) / 256, 256, 0, stream>>>(
            dB1, Bf, meanv, 1.0f / (float)NDIM);

        // G[H,H] = B^T[H,M] @ dB1[M,H]  (TN, K=M)
        gemm_bf16_wmma<true, false><<<dim3(HDIM / 128, HDIM / 128), 256, 0, stream>>>(
            Bf, dB1, G, MDIM, HDIM, HDIM, HDIM, 1.0f);

        // P[M,H] = B[M,H] @ G[H,H]  (NN)
        gemm_bf16_wmma<false, false><<<dim3(HDIM / 128, MDIM / 128), 256, 0, stream>>>(
            Bf, G, P, HDIM, HDIM, HDIM, HDIM, 1.0f);

        // Langevin update + row l2norm
        update_kernel<<<MDIM, 256, 0, stream>>>(
            Bf, dB1, P, noise + (size_t)i * MDIM * HDIM, c, sn,
            1.0f / (float)MDIM);
    }

    // logit_neg[N,M] = qn @ B^T / TEMP, strided into d_out cols 1..M  (NT)
    gemm_bf16_wmma<false, true><<<dim3(MDIM / 128, NDIM / 128), 256, 0, stream>>>(
        qn, Bf, out + 1, HDIM, HDIM, HDIM, MDIM + 1, TEMP_INV);
}
